// DGCNN_31241592111462
// MI455X (gfx1250) — matrix operand order repaired and checked by reading, not compiled, not run on hardware
//
#include <hip/hip_runtime.h>
#include <hip/hip_bf16.h>

// ---------------- problem constants ----------------
#define GG     256        // graphs
#define NPG    128        // nodes per graph
#define HID    256
#define NN     32768      // nodes
#define EE     524288     // edges
#define TD     769        // total concat dim
#define KTOP   30
#define KD     23070      // KTOP*TD
#define C1LEN  23066
#define P1LEN  11533
#define C2LEN  11529
#define FLATD  368928     // 32*C2LEN = 216 * 1708 (exact split-K)
#define KSPLIT 216
#define KCHUNK 1708       // 427 k4-steps per chunk; 216*1708 == FLATD exactly
#define KSTEPS 427

typedef float v2f  __attribute__((ext_vector_type(2)));
typedef float v8f  __attribute__((ext_vector_type(8)));
typedef __bf16 v16bf __attribute__((ext_vector_type(16)));

union ABf { uint4 u4[2]; v16bf v; };

__device__ inline unsigned short f2bf(float f) {
    union { float f; unsigned int u; } v; v.f = f;
    unsigned int u = v.u;
    unsigned int r = u + 0x7FFFu + ((u >> 16) & 1u);   // round-to-nearest-even
    return (unsigned short)(r >> 16);
}

// ---------------- degree / normalization ----------------
__global__ void k_deg_init(float* deg) {
    int i = blockIdx.x * 256 + threadIdx.x;
    deg[i] = 1.0f;                                      // self loop
}
__global__ void k_deg_count(const int* dst, float* deg) {
    int e = blockIdx.x * 256 + threadIdx.x;
    atomicAdd(&deg[dst[e]], 1.0f);
}
__global__ void k_norm(const float* deg, float* dinv, float* snorm) {
    int i = blockIdx.x * 256 + threadIdx.x;
    float d = deg[i];
    dinv[i]  = rsqrtf(d);
    snorm[i] = 1.0f / d;
}

// ---------------- embedding gather (f32 + bf16 shadow) ----------------
__global__ void k_embed(const int* z, const float* emb, float* x, unsigned short* xb) {
    int i = blockIdx.x * 256 + threadIdx.x;            // i in [0, NN*HID)
    int node = i >> 8, c = i & 255;
    float v = emb[(size_t)z[node] * HID + c];
    x[i]  = v;
    xb[i] = f2bf(v);
}

// ---------------- transpose+convert GCN weight: Wbt[n][k] = bf16(W[k][n]) ----------------
__global__ void k_convW(const float* W, unsigned short* wbt) {
    int i = blockIdx.x * 256 + threadIdx.x;            // i in [0, HID*HID)
    int n = i >> 8, k = i & 255;
    wbt[(size_t)n * HID + k] = f2bf(W[(size_t)k * HID + n]);
}

// ---------------- GCN GEMM: xw = xbf16 @ Wbt^T  (bf16 WMMA, f32 acc) ----------------
__global__ void __launch_bounds__(256) k_gemm_gcn(const unsigned short* xb,
                                                  const unsigned short* wbt, float* xw) {
    int wid  = blockIdx.x * 8 + (threadIdx.x >> 5);    // 32768 waves = 2048 Mtiles x 16 Ntiles
    int lane = threadIdx.x & 31, half = lane >> 4, lm = lane & 15;
    int tm = wid >> 4, tn = wid & 15;
    const unsigned short* arow = xb  + (size_t)(tm * 16 + lm) * HID;
    const unsigned short* brow = wbt + (size_t)(tn * 16 + lm) * HID;
    v8f acc = {0.f,0.f,0.f,0.f,0.f,0.f,0.f,0.f};
    for (int k0 = 0; k0 < HID; k0 += 32) {
        ABf A, B;
        A.u4[0] = *(const uint4*)(arow + k0 + half * 8);        // v0..3: K = k0 + half*8 + 0..7
        A.u4[1] = *(const uint4*)(arow + k0 + 16 + half * 8);   // v4..7: K = k0+16 + half*8 + 0..7
        B.u4[0] = *(const uint4*)(brow + k0 + half * 8);        // K = k0 + half*8 + 0..7
        B.u4[1] = *(const uint4*)(brow + k0 + 16 + half * 8);   // K = k0+16 + half*8 + 0..7
        acc = __builtin_amdgcn_wmma_f32_16x16x32_bf16(false, A.v, false, B.v,
                                                      (short)0, acc, false, false);
    }
    float* crow = xw + (size_t)(tm * 16 + half * 8) * HID + tn * 16 + lm;
    #pragma unroll
    for (int r = 0; r < 8; r++) crow[(size_t)r * HID] = acc[r];
}

// ---------------- GCN aggregation ----------------
__global__ void k_agg_init(const float* xw, const float* snorm, const float* b, float* agg) {
    int i = blockIdx.x * 256 + threadIdx.x;
    agg[i] = snorm[i >> 8] * xw[i] + b[i & 255];
}
__global__ void k_scatter(const int* src, const int* dst, const float* dinv,
                          const float* xw, float* agg) {
    int e = blockIdx.x * 8 + (threadIdx.x >> 5);       // one wave per edge
    int lane = threadIdx.x & 31;
    int s = src[e], d = dst[e];
    float nrm = dinv[s] * dinv[d];
    const float* xs = xw + (size_t)s * HID;
    float* ad = agg + (size_t)d * HID;
    #pragma unroll
    for (int it = 0; it < 8; it++) {
        int c = lane + it * 32;
        atomicAdd(&ad[c], nrm * xs[c]);
    }
}
__global__ void k_tanh(const float* agg, float* x, unsigned short* xb, float* h, int loff) {
    int i = blockIdx.x * 256 + threadIdx.x;
    int node = i >> 8, c = i & 255;
    float v = tanhf(agg[i]);
    x[i]  = v;
    xb[i] = f2bf(v);
    h[(size_t)node * TD + loff + c] = v;
}

// ---------------- last GCN layer (HID -> 1) ----------------
__global__ void k_last(const float* x, const float* wl, float* xwL) {
    int wave = threadIdx.x >> 5, lane = threadIdx.x & 31;
    int node = blockIdx.x * 8 + wave;
    const float* row = x + (size_t)node * HID;
    float s = 0.f;
    #pragma unroll
    for (int it = 0; it < 8; it++) s += row[lane + it * 32] * wl[lane + it * 32];
    #pragma unroll
    for (int m = 16; m > 0; m >>= 1) s += __shfl_xor(s, m, 32);
    if (lane == 0) xwL[node] = s;
}
__global__ void k_aggL_init(const float* xwL, const float* snorm, const float* bL, float* aggL) {
    int i = blockIdx.x * 256 + threadIdx.x;
    aggL[i] = snorm[i] * xwL[i] + bL[0];
}
__global__ void k_scatterL(const int* src, const int* dst, const float* dinv,
                           const float* xwL, float* aggL) {
    int e = blockIdx.x * 256 + threadIdx.x;
    int s = src[e], d = dst[e];
    atomicAdd(&aggL[d], dinv[s] * dinv[d] * xwL[s]);
}
__global__ void k_tanhL(const float* aggL, float* h) {
    int i = blockIdx.x * 256 + threadIdx.x;
    h[(size_t)i * TD + (TD - 1)] = tanhf(aggL[i]);
}

// ---------------- global_sort_pool: bitonic sort 128 keys, gather top-30 ----------------
__global__ void k_sortpool(const float* h, float* p) {
    __shared__ float sk[NPG];
    __shared__ int   si[NPG];
    int g = blockIdx.x, t = threadIdx.x;
    sk[t] = h[(size_t)(g * NPG + t) * TD + (TD - 1)];
    si[t] = t;
    __syncthreads();
    for (int k = 2; k <= NPG; k <<= 1)
        for (int j = k >> 1; j > 0; j >>= 1) {
            int ixj = t ^ j;
            if (ixj > t) {
                bool up = ((t & k) == 0);
                float ka = sk[t], kb = sk[ixj];
                int ia = si[t], ib = si[ixj];
                // "before" = larger key first; ties -> smaller original index (stable argsort)
                bool b_first = (kb > ka) || (kb == ka && ib < ia);
                bool a_first = (ka > kb) || (ka == kb && ia < ib);
                if (up ? b_first : a_first) { sk[t] = kb; sk[ixj] = ka; si[t] = ib; si[ixj] = ia; }
            }
            __syncthreads();
        }
    for (int li = t; li < KD; li += NPG) {
        int r = li / TD, c = li - r * TD;
        p[(size_t)g * KD + li] = h[(size_t)(g * NPG + si[r]) * TD + c];
    }
}

// ---------------- fused conv1 + relu + maxpool2 + conv2 + relu ----------------
__global__ void __launch_bounds__(256) k_conv(const float* p, const float* w1, const float* b1,
                                              const float* w2, const float* b2, float* flatb) {
    __shared__ float sp[140];
    __shared__ float pooled[16 * 68];
    __shared__ float w2s[32 * 16 * 5];
    __shared__ float w1s[80];
    __shared__ float b1s[16];
    __shared__ float b2s[32];
    int g = blockIdx.y, t0 = blockIdx.x * 64, tid = threadIdx.x;
    const float* pg = p + (size_t)g * KD;
    for (int i = tid; i < 2560; i += 256) w2s[i] = w2[i];
    if (tid < 80) w1s[tid] = w1[tid];
    if (tid < 16) b1s[tid] = b1[tid];
    if (tid < 32) b2s[tid] = b2[tid];
    for (int i = tid; i < 140; i += 256) {
        int gi = 2 * t0 + i;
        sp[i] = (gi < KD) ? pg[gi] : 0.f;
    }
    __syncthreads();
    // conv1 (k=5) + relu + maxpool(2) computed straight into `pooled`
    for (int task = tid; task < 16 * 68; task += 256) {
        int ch = task / 68, lu = task - ch * 68;
        const float* wch = &w1s[ch * 5];
        float c1a = b1s[ch], c1b = b1s[ch];
        int lv = 2 * lu;
        #pragma unroll
        for (int i = 0; i < 5; i++) { c1a += sp[lv + i] * wch[i]; c1b += sp[lv + 1 + i] * wch[i]; }
        c1a = fmaxf(c1a, 0.f); c1b = fmaxf(c1b, 0.f);
        pooled[ch * 68 + lu] = fmaxf(c1a, c1b);
    }
    __syncthreads();
    // conv2 (16->32, k=5) + relu
    int oc = tid >> 3;
    for (int j = 0; j < 8; j++) {
        int lt = (tid & 7) + j * 8;
        int t = t0 + lt;
        if (t >= C2LEN) continue;
        float acc = b2s[oc];
        #pragma unroll
        for (int ic = 0; ic < 16; ic++) {
            const float* pl = &pooled[ic * 68 + lt];
            const float* wl = &w2s[oc * 80 + ic * 5];
            #pragma unroll
            for (int i = 0; i < 5; i++) acc += pl[i] * wl[i];
        }
        flatb[(size_t)g * FLATD + (size_t)oc * C2LEN + t] = fmaxf(acc, 0.f);
    }
}

// ---- async staging for lin1: per thread one b128 (A row) + one b64 (B half-row) ----
// LDS byte offset = low 32 bits of the generic pointer (flat-LDS aperture mapping, ISA 10.2)
__device__ __forceinline__ void lin1_prefetch(const float* flatb, const float* l1w,
                                              float* Ad, float* Bd, int tid, int k0) {
    const float* ga = flatb + (size_t)tid * FLATD + k0;            // 16B aligned
    unsigned la = (unsigned)(size_t)(Ad + tid * 4);
    asm volatile("global_load_async_to_lds_b128 %0, %1, off"
                 :: "v"(la), "v"(ga) : "memory");
    int row = tid >> 1, h8 = tid & 1;
    const float* gb = l1w + (size_t)row * FLATD + k0 + h8 * 2;     // 8B aligned
    unsigned lb = (unsigned)(size_t)(Bd + row * 4 + h8 * 2);
    asm volatile("global_load_async_to_lds_b64 %0, %1, off"
                 :: "v"(lb), "v"(gb) : "memory");
}

// ---------------- lin1: split-K fp32 WMMA, double-buffered async LDS staging ----------------
__global__ void __launch_bounds__(256, 1) k_lin1(const float* flatb, const float* l1w,
                                                 float* fpart) {
    __shared__ __align__(16) float As[2][256 * 4];   // A tile: 256 graphs x 4 k
    __shared__ __align__(16) float Bs[2][128 * 4];   // B tile (transposed): 128 n x 4 k
    int s = blockIdx.x;
    int tid = threadIdx.x;
    int wv = tid >> 5, lane = tid & 31, half = lane >> 4, lm = lane & 15;
    int n = wv * 16 + lm;
    int kk = half * 2;
    int kbase = s * KCHUNK;
    v8f acc[16];
    const v8f Z = {0.f,0.f,0.f,0.f,0.f,0.f,0.f,0.f};
    #pragma unroll
    for (int i = 0; i < 16; i++) acc[i] = Z;

    lin1_prefetch(flatb, l1w, As[0], Bs[0], tid, kbase);           // prologue -> buf 0

    for (int step = 0; step < KSTEPS; step++) {
        int cur = step & 1;
        // prefetch next tile into the other buffer (wraps harmlessly on the last step)
        int knext = (step + 1 < KSTEPS) ? kbase + (step + 1) * 4 : kbase;
        lin1_prefetch(flatb, l1w, As[1 - cur], Bs[1 - cur], tid, knext);
        // 2 outstanding = the pair just issued; the current buffer's pair has completed
        asm volatile("s_wait_asynccnt 0x2" ::: "memory");
        __syncthreads();                                           // all waves' fills visible
        v2f bfrag = { Bs[cur][n * 4 + kk], Bs[cur][n * 4 + kk + 1] };   // B[K=half*2+v, N=n]
        #pragma unroll
        for (int mt = 0; mt < 16; mt++) {
            int m = mt * 16 + lm;
            v2f afrag = { As[cur][m * 4 + kk], As[cur][m * 4 + kk + 1] }; // A[M=m, K=half*2+v]
            acc[mt] = __builtin_amdgcn_wmma_f32_16x16x4_f32(false, afrag, false, bfrag,
                                                            (short)0, acc[mt], false, false);
        }
        __syncthreads();                                           // done reading cur buffer
    }
    // write partials: fpart[s][m][n]
    float* base = fpart + (size_t)s * (GG * 128);
    #pragma unroll
    for (int mt = 0; mt < 16; mt++) {
        #pragma unroll
        for (int r = 0; r < 8; r++) {
            int m = mt * 16 + half * 8 + r;
            base[(size_t)m * 128 + n] = acc[mt][r];
        }
    }
}

// ---------------- reduce split-K partials + bias + relu + lin2 ----------------
__global__ void k_lin2(const float* fpart, const float* b1, const float* w2,
                       const float* b2, float* out) {
    __shared__ float red[128];
    int g = blockIdx.x, n = threadIdx.x;
    float acc = 0.f;
    for (int s = 0; s < KSPLIT; s++) acc += fpart[(size_t)s * (GG * 128) + g * 128 + n];
    float v = fmaxf(acc + b1[n], 0.f);
    red[n] = v * w2[n];
    __syncthreads();
    for (int off = 64; off > 0; off >>= 1) {
        if (n < off) red[n] += red[n + off];
        __syncthreads();
    }
    if (n == 0) out[g] = red[0] + b2[0];
}

// ---------------- host orchestration ----------------
extern "C" void kernel_launch(void* const* d_in, const int* in_sizes, int n_in,
                              void* d_out, int out_size, void* d_ws, size_t ws_size,
                              hipStream_t stream) {
    const int*   z     = (const int*)d_in[0];
    const int*   ei    = (const int*)d_in[1];
    const int*   src   = ei;
    const int*   dst   = ei + EE;
    const float* emb   = (const float*)d_in[3];
    const float* gcn_w = (const float*)d_in[4];
    const float* gcn_b = (const float*)d_in[5];
    const float* gwL   = (const float*)d_in[6];
    const float* gbL   = (const float*)d_in[7];
    const float* c1w   = (const float*)d_in[8];
    const float* c1b   = (const float*)d_in[9];
    const float* c2w   = (const float*)d_in[10];
    const float* c2b   = (const float*)d_in[11];
    const float* l1w   = (const float*)d_in[12];
    const float* l1b   = (const float*)d_in[13];
    const float* l2w   = (const float*)d_in[14];
    const float* l2b   = (const float*)d_in[15];
    float* out = (float*)d_out;

    char* wsp = (char*)d_ws;
    auto take = [&](size_t nbytes) {
        char* r = wsp;
        wsp += (nbytes + 255) & ~(size_t)255;
        return r;
    };
    float*          deg   = (float*)take((size_t)NN * 4);
    float*          dinv  = (float*)take((size_t)NN * 4);
    float*          snorm = (float*)take((size_t)NN * 4);
    float*          x     = (float*)take((size_t)NN * HID * 4);
    unsigned short* xb    = (unsigned short*)take((size_t)NN * HID * 2);
    float*          xw    = (float*)take((size_t)NN * HID * 4);
    float*          agg   = (float*)take((size_t)NN * HID * 4);
    unsigned short* wbt   = (unsigned short*)take((size_t)HID * HID * 2);
    float*          h     = (float*)take((size_t)NN * TD * 4);
    float*          xwL   = (float*)take((size_t)NN * 4);
    float*          aggL  = (float*)take((size_t)NN * 4);
    float*          pbuf  = (float*)take((size_t)GG * KD * 4);
    float*          flatb = (float*)take((size_t)GG * FLATD * 4);
    float*          fpart = (float*)take((size_t)KSPLIT * GG * 128 * 4);

    k_deg_init<<<NN / 256, 256, 0, stream>>>(deg);
    k_deg_count<<<EE / 256, 256, 0, stream>>>(dst, deg);
    k_norm<<<NN / 256, 256, 0, stream>>>(deg, dinv, snorm);
    k_embed<<<(NN * HID) / 256, 256, 0, stream>>>(z, emb, x, xb);

    for (int l = 0; l < 3; l++) {
        k_convW<<<(HID * HID) / 256, 256, 0, stream>>>(gcn_w + (size_t)l * HID * HID, wbt);
        k_gemm_gcn<<<4096, 256, 0, stream>>>(xb, wbt, xw);
        k_agg_init<<<(NN * HID) / 256, 256, 0, stream>>>(xw, snorm, gcn_b + (size_t)l * HID, agg);
        k_scatter<<<EE / 8, 256, 0, stream>>>(src, dst, dinv, xw, agg);
        k_tanh<<<(NN * HID) / 256, 256, 0, stream>>>(agg, x, xb, h, l * HID);
    }

    k_last<<<NN / 8, 256, 0, stream>>>(x, gwL, xwL);
    k_aggL_init<<<NN / 256, 256, 0, stream>>>(xwL, snorm, gbL, aggL);
    k_scatterL<<<EE / 256, 256, 0, stream>>>(src, dst, dinv, xwL, aggL);
    k_tanhL<<<NN / 256, 256, 0, stream>>>(aggL, h);

    k_sortpool<<<GG, NPG, 0, stream>>>(h, pbuf);

    dim3 cgrid((C2LEN + 63) / 64, GG);
    k_conv<<<cgrid, 256, 0, stream>>>(pbuf, c1w, c1b, c2w, c2b, flatb);

    k_lin1<<<KSPLIT, 256, 0, stream>>>(flatb, l1w, fpart);
    k_lin2<<<GG, 128, 0, stream>>>(fpart, l1b, l2w, l2b, out);
}
